// SiglipAttention_65008624992257
// MI455X (gfx1250) — matrix operand order repaired and checked by reading, not compiled, not run on hardware
//
#include <hip/hip_runtime.h>
#include <hip/hip_bf16.h>

// ---------------- problem constants ----------------
#define BQ   8
#define SEQ  1024
#define DIM  1152
#define NH   16
#define HD   72
#define HDP  96            // padded head dim (3 x 32 for WMMA K-steps)
#define MROWS (BQ*SEQ)     // 8192
#define KT_PROJ (DIM/32)   // 36

static __device__ __constant__ const float SCALE_F = 0.11785113019775793f; // 72^-0.5

typedef __attribute__((ext_vector_type(16))) __bf16 v16bf;
typedef __attribute__((ext_vector_type(8)))  float  v8f;
typedef __attribute__((ext_vector_type(4)))  unsigned v4u;
typedef __attribute__((ext_vector_type(8)))  int      v8i;
typedef __attribute__((ext_vector_type(4)))  int      v4i;

union BFrag { unsigned u[8]; v16bf v; };

__device__ inline v8f vzero() {
  v8f z;
#pragma unroll
  for (int i = 0; i < 8; ++i) z[i] = 0.0f;
  return z;
}

// round-to-nearest-even f32 -> bf16 pack helpers
__device__ inline unsigned pk2(float a, float b) {
  unsigned ua = __float_as_uint(a); ua = (ua + 0x7FFFu + ((ua >> 16) & 1u)) >> 16;
  unsigned ub = __float_as_uint(b); ub = (ub + 0x7FFFu + ((ub >> 16) & 1u)) >> 16;
  return ua | (ub << 16);
}
__device__ inline unsigned short bf1(float a) {
  unsigned ua = __float_as_uint(a);
  return (unsigned short)((ua + 0x7FFFu + ((ua >> 16) & 1u)) >> 16);
}

__device__ inline v8f wmma_bf16(const BFrag& a, const BFrag& b, v8f c) {
  return __builtin_amdgcn_wmma_f32_16x16x32_bf16(
      /*neg_a=*/false, a.v, /*neg_b=*/false, b.v,
      /*c_mod=*/(short)0, c, /*reuse_a=*/false, /*reuse_b=*/false);
}

// ---------------- Tensor Data Mover helpers ----------------
// generic shared pointer low 32 bits == LDS byte offset (flat LDS aperture mapping)
__device__ inline unsigned lds_offset(const void* p) {
  return (unsigned)(unsigned long long)(uintptr_t)p;
}

__device__ inline void tdm_issue(v4u g0, v8i g1) {
  v4i z4 = {0, 0, 0, 0};
#if defined(__clang_major__) && (__clang_major__ >= 23)
  v8i z8 = {0, 0, 0, 0, 0, 0, 0, 0};
  __builtin_amdgcn_tensor_load_to_lds(g0, g1, z4, z4, z8, 0);
#else
  __builtin_amdgcn_tensor_load_to_lds(g0, g1, z4, z4, 0);
#endif
}

// 2D tile load: tile_d1 rows x tile_d0 elements, element size / LDS padding
// encoded in w0flags (D# group1 word0: data_size[17:16], pad_enable[20],
// pad_interval[24:22], pad_amount[31:25]).
__device__ inline void tdm_load_2d(unsigned ldsOff, const void* gaddr,
                                   unsigned w0flags,
                                   unsigned tensor_d0, unsigned tensor_d1,
                                   unsigned tile_d0, unsigned tile_d1,
                                   unsigned stride0) {
  unsigned long long ga = (unsigned long long)(uintptr_t)gaddr;
  v4u g0;
  g0[0] = 1u;                                              // count=1, user mode
  g0[1] = ldsOff;                                          // lds_addr (bytes)
  g0[2] = (unsigned)(ga & 0xFFFFFFFFu);                    // global_addr[31:0]
  g0[3] = (unsigned)((ga >> 32) & 0x1FFFFFFu) | (2u << 30); // addr[56:32] | type=2
  v8i g1;
  g1[0] = (int)w0flags;                                    // mask=0 | data_size | pad cfg
  g1[1] = (int)((tensor_d0 & 0xFFFFu) << 16);              // tensor_dim0[15:0] @ bit48
  g1[2] = (int)((tensor_d0 >> 16) | ((tensor_d1 & 0xFFFFu) << 16));
  g1[3] = (int)((tensor_d1 >> 16) | (tile_d0 << 16));      // tile_dim0 @ bit112
  g1[4] = (int)(tile_d1 & 0xFFFFu);                        // tile_dim1 (tile_dim2=0)
  g1[5] = (int)stride0;                                    // tensor_dim0_stride[31:0]
  g1[6] = 0;
  g1[7] = 0;
  tdm_issue(g0, g1);
}

// ---------------- small conversion / init kernels ----------------
__global__ void cvt_pairs(const float* __restrict__ in, unsigned* __restrict__ out, int npair) {
  for (int i = blockIdx.x * blockDim.x + threadIdx.x; i < npair; i += gridDim.x * blockDim.x) {
    float2 f = *(const float2*)(in + (size_t)i * 2);
    out[i] = pk2(f.x, f.y);
  }
}

// W is [out=1152][in=1152]; B operand B[k][n] = W[n][k], pair-packed over k:
// out_pk[p*1152 + n] = { lo = W[n][2p], hi = W[n][2p+1] }
__global__ void cvt_weight_pk(const float* __restrict__ W, unsigned* __restrict__ out) {
  const int n = (DIM / 2) * DIM; // 576*1152
  for (int i = blockIdx.x * blockDim.x + threadIdx.x; i < n; i += gridDim.x * blockDim.x) {
    int col = i % DIM;            // n (output feature)
    int p   = i / DIM;            // k-pair
    float lo = W[(size_t)col * DIM + 2 * p];
    float hi = W[(size_t)col * DIM + 2 * p + 1];
    out[(size_t)p * DIM + col] = pk2(lo, hi);
  }
}

__global__ void zero_u32(unsigned* __restrict__ p, size_t n) {
  for (size_t i = blockIdx.x * (size_t)blockDim.x + threadIdx.x; i < n;
       i += (size_t)gridDim.x * blockDim.x) p[i] = 0u;
}

// ---------------- projection GEMM (QKV + O), TDM double-buffered ----------------
// C[8192 x 1152] = A_bf16 @ Bpk + bias ; 128x128 block tile, 8 waves (2x4),
// each wave 4x2 tiles of 16x16. K-step 32 (pair-packed B rows 16 per step).
// Wave 0 issues TDM loads for stage kt+1 while all waves run WMMAs on stage kt.
// MODE 0: Q -> qp  [head][tok][96] bf16, scaled by SCALE
// MODE 1: K -> kT  [head][48][1024] u32 pair-packed over head-dim
// MODE 2: V -> vp  [head][512][96] u32 pair-packed over tokens
// MODE 3: O -> f32 d_out [8192][1152]
template <int MODE>
__global__ __launch_bounds__(256) void proj_gemm(
    const unsigned short* __restrict__ A,   // [8192][1152] bf16
    const unsigned* __restrict__ Bpk,       // [576][1152] u32
    const float* __restrict__ bias,         // [1152]
    unsigned short* __restrict__ out_b,
    float* __restrict__ out_f) {
  __shared__ unsigned short As[2][128 * 40];  // stride 40 bf16 (TDM pad: 16B per 64B row)
  __shared__ unsigned Bs[2][16 * 132];        // stride 132 u32 (TDM pad: 16B per 512B row)

  const int tid  = threadIdx.x;
  const int wave = tid >> 5, lane = tid & 31;
  const int lr = lane & 15, lh = lane >> 4;
  const int wr = wave >> 2, wc = wave & 3;             // 2x4 wave grid
  const int rowBase = blockIdx.y * 128;
  const int colBase = blockIdx.x * 128;

  // w0flags: A tile bf16 (data_size=1), pad 4 dwords every 16 dwords
  const unsigned WA = (1u << 16) | (1u << 20) | (3u << 22) | (3u << 25);
  // w0flags: B tile u32 (data_size=2), pad 4 dwords every 128 dwords
  const unsigned WB = (2u << 16) | (1u << 20) | (6u << 22) | (3u << 25);

  v8f acc[4][2];
#pragma unroll
  for (int i = 0; i < 4; ++i)
#pragma unroll
    for (int j = 0; j < 2; ++j) acc[i][j] = vzero();

  if (wave == 0) {  // preload stage 0
    tdm_load_2d(lds_offset(As[0]), A + (size_t)rowBase * DIM, WA,
                DIM, MROWS, 32, 128, DIM);
    tdm_load_2d(lds_offset(Bs[0]), Bpk + (size_t)0 * DIM + colBase, WB,
                DIM, DIM / 2, 128, 16, DIM);
  }

  for (int kt = 0; kt < KT_PROJ; ++kt) {
    __syncthreads();  // everyone done reading the buffer we are about to refill
    if (wave == 0) {
      if (kt + 1 < KT_PROJ) {
        const int nb = (kt + 1) & 1;
        tdm_load_2d(lds_offset(As[nb]),
                    A + (size_t)rowBase * DIM + (kt + 1) * 32, WA,
                    DIM, MROWS, 32, 128, DIM);
        tdm_load_2d(lds_offset(Bs[nb]),
                    Bpk + (size_t)((kt + 1) * 16) * DIM + colBase, WB,
                    DIM, DIM / 2, 128, 16, DIM);
        __builtin_amdgcn_s_wait_tensorcnt(2);  // current stage's 2 loads done
      } else {
        __builtin_amdgcn_s_wait_tensorcnt(0);
      }
    }
    __syncthreads();

    const unsigned* Abuf = (const unsigned*)As[kt & 1];
    const unsigned* Bbuf = Bs[kt & 1];

    BFrag a[4], b[2];
#pragma unroll
    for (int rt = 0; rt < 4; ++rt) {
      int row = wr * 64 + rt * 16 + lr;
#pragma unroll
      for (int j = 0; j < 8; ++j) {
        int kk = lh * 16 + j * 2;
        a[rt].u[j] = Abuf[row * 20 + (kk >> 1)];
      }
    }
#pragma unroll
    for (int ct = 0; ct < 2; ++ct) {
      int col = wc * 32 + ct * 16 + lr;
#pragma unroll
      for (int j = 0; j < 8; ++j) b[ct].u[j] = Bbuf[(lh * 8 + j) * 132 + col];
    }
#pragma unroll
    for (int rt = 0; rt < 4; ++rt)
#pragma unroll
      for (int ct = 0; ct < 2; ++ct) acc[rt][ct] = wmma_bf16(a[rt], b[ct], acc[rt][ct]);
  }

  // epilogue
#pragma unroll
  for (int rt = 0; rt < 4; ++rt) {
#pragma unroll
    for (int ct = 0; ct < 2; ++ct) {
      const int gcol = colBase + wc * 32 + ct * 16 + lr;
      const float bv = bias[gcol];
#pragma unroll
      for (int i = 0; i < 8; ++i) {
        const int grow = rowBase + wr * 64 + rt * 16 + lh * 8 + i;
        const float v = acc[rt][ct][i] + bv;
        if (MODE == 3) {
          out_f[(size_t)grow * DIM + gcol] = v;
        } else {
          const int bq = grow >> 10, tok = grow & 1023;
          const int h = gcol / HD, dd = gcol - h * HD;
          const int head = bq * NH + h;
          if (MODE == 0) {
            out_b[((size_t)head * SEQ + tok) * HDP + dd] = bf1(v * SCALE_F);
          } else if (MODE == 1) {  // kT pair-packed over head-dim
            size_t u = (size_t)head * (HDP / 2) * SEQ + (size_t)(dd >> 1) * SEQ + tok;
            out_b[u * 2 + (dd & 1)] = bf1(v);
          } else {                 // vp pair-packed over tokens
            size_t u = (size_t)head * (SEQ / 2) * HDP + (size_t)(tok >> 1) * HDP + dd;
            out_b[u * 2 + (tok & 1)] = bf1(v);
          }
        }
      }
    }
  }
}

// ---------------- scores GEMM: probs_raw = q_scaled @ k^T ----------------
__global__ __launch_bounds__(256) void scores_gemm(
    const unsigned short* __restrict__ qp,  // [128][1024][96] bf16
    const unsigned* __restrict__ kT,        // [128][48][1024] u32
    float* __restrict__ probs) {            // [128][1024][1024] f32
  __shared__ unsigned short As[128 * 40];
  __shared__ unsigned Bs[16 * 132];

  const int tid = threadIdx.x;
  const int wave = tid >> 5, lane = tid & 31;
  const int lr = lane & 15, lh = lane >> 4;
  const int wr = wave >> 2, wc = wave & 3;
  const int bh = blockIdx.z;
  const int rowBase = blockIdx.y * 128;
  const int colBase = blockIdx.x * 128;

  const unsigned short* Ah = qp + (size_t)bh * SEQ * HDP;
  const unsigned* Bh = kT + (size_t)bh * (HDP / 2) * SEQ;
  float* Ch = probs + (size_t)bh * SEQ * SEQ;

  v8f acc[4][2];
#pragma unroll
  for (int i = 0; i < 4; ++i)
#pragma unroll
    for (int j = 0; j < 2; ++j) acc[i][j] = vzero();

  const unsigned* Alds32 = (const unsigned*)As;

  for (int kt = 0; kt < HDP / 32; ++kt) {  // 3 steps
#pragma unroll
    for (int i = 0; i < 2; ++i) {
      int e = tid + i * 256;
      int r = e >> 2, q = e & 3;
      *(uint4*)(As + r * 40 + q * 8) =
          *(const uint4*)(Ah + (size_t)(rowBase + r) * HDP + kt * 32 + q * 8);
    }
#pragma unroll
    for (int i = 0; i < 2; ++i) {
      int e = tid + i * 256;
      int r = e >> 5, q = e & 31;
      *(uint4*)(Bs + r * 132 + q * 4) =
          *(const uint4*)(Bh + (size_t)(kt * 16 + r) * SEQ + colBase + q * 4);
    }
    __syncthreads();

    BFrag a[4], b[2];
#pragma unroll
    for (int rt = 0; rt < 4; ++rt) {
      int row = wr * 64 + rt * 16 + lr;
#pragma unroll
      for (int j = 0; j < 8; ++j) {
        int kk = lh * 16 + j * 2;
        a[rt].u[j] = Alds32[row * 20 + (kk >> 1)];
      }
    }
#pragma unroll
    for (int ct = 0; ct < 2; ++ct) {
      int col = wc * 32 + ct * 16 + lr;
#pragma unroll
      for (int j = 0; j < 8; ++j) b[ct].u[j] = Bs[(lh * 8 + j) * 132 + col];
    }
#pragma unroll
    for (int rt = 0; rt < 4; ++rt)
#pragma unroll
      for (int ct = 0; ct < 2; ++ct) acc[rt][ct] = wmma_bf16(a[rt], b[ct], acc[rt][ct]);
    __syncthreads();
  }

#pragma unroll
  for (int rt = 0; rt < 4; ++rt)
#pragma unroll
    for (int ct = 0; ct < 2; ++ct) {
      const int gcol = colBase + wc * 32 + ct * 16 + lr;
#pragma unroll
      for (int i = 0; i < 8; ++i) {
        const int grow = rowBase + wr * 64 + rt * 16 + lh * 8 + i;
        Ch[(size_t)grow * SEQ + gcol] = acc[rt][ct][i];
      }
    }
}

// ---------------- in-place row softmax over 1024 ----------------
__global__ __launch_bounds__(256) void softmax_rows(float* __restrict__ probs) {
  __shared__ float red[256];
  float* p = probs + (size_t)blockIdx.x * SEQ;
  const int t = threadIdx.x;
  float4 x = ((const float4*)p)[t];

  float m = fmaxf(fmaxf(x.x, x.y), fmaxf(x.z, x.w));
  red[t] = m;
  __syncthreads();
  for (int s = 128; s > 0; s >>= 1) {
    if (t < s) red[t] = fmaxf(red[t], red[t + s]);
    __syncthreads();
  }
  m = red[0];
  __syncthreads();

  x.x = __expf(x.x - m); x.y = __expf(x.y - m);
  x.z = __expf(x.z - m); x.w = __expf(x.w - m);
  red[t] = x.x + x.y + x.z + x.w;
  __syncthreads();
  for (int s = 128; s > 0; s >>= 1) {
    if (t < s) red[t] += red[t + s];
    __syncthreads();
  }
  const float inv = 1.0f / red[0];
  x.x *= inv; x.y *= inv; x.z *= inv; x.w *= inv;
  ((float4*)p)[t] = x;
}

// ---------------- probs @ V ----------------
// per (b,h): C[1024x96] = probs[1024x1024] @ V[1024x96]; block 128 rows x 96 cols,
// 8 waves as 4x2, wave tile 32x48 = 2x3 tiles. probs converted f32->bf16 into LDS.
__global__ __launch_bounds__(256) void attnv_gemm(
    const float* __restrict__ probs,   // [128][1024][1024]
    const unsigned* __restrict__ vp,   // [128][512][96] u32
    unsigned short* __restrict__ otmp) // [8192][1152] bf16
{
  __shared__ unsigned short As[128 * 40];
  __shared__ unsigned Bs[16 * 96];

  const int tid = threadIdx.x;
  const int wave = tid >> 5, lane = tid & 31;
  const int lr = lane & 15, lh = lane >> 4;
  const int wr = wave >> 1, wc = wave & 1;  // 4x2 wave grid
  const int bh = blockIdx.y;
  const int rowBase = blockIdx.x * 128;
  const int bq = bh >> 4, h = bh & 15;

  const float* Ph = probs + (size_t)bh * SEQ * SEQ;
  const unsigned* Vh = vp + (size_t)bh * (SEQ / 2) * HDP;

  v8f acc[2][3];
#pragma unroll
  for (int i = 0; i < 2; ++i)
#pragma unroll
    for (int j = 0; j < 3; ++j) acc[i][j] = vzero();

  unsigned* Alds32 = (unsigned*)As;

  for (int kt = 0; kt < SEQ / 32; ++kt) {  // 32 steps
    // stage + convert probs tile 128x32 f32 -> bf16 pairs
#pragma unroll
    for (int i = 0; i < 8; ++i) {
      int e = tid + i * 256;
      int r = e >> 4, q = e & 15;
      float2 f = *(const float2*)(Ph + (size_t)(rowBase + r) * SEQ + kt * 32 + q * 2);
      Alds32[r * 20 + q] = pk2(f.x, f.y);
    }
    // stage V tile: pk rows kt*16..+15, 96 cols (384 uint4)
    for (int e = tid; e < 384; e += 256) {
      int r = e / 24, q = e - r * 24;
      *(uint4*)(Bs + r * 96 + q * 4) =
          *(const uint4*)(Vh + (size_t)(kt * 16 + r) * HDP + q * 4);
    }
    if (kt + 1 < SEQ / 32)
      __builtin_prefetch(Ph + (size_t)(rowBase + (tid >> 1)) * SEQ + (kt + 1) * 32, 0, 1);
    __syncthreads();

    BFrag a[2], b[3];
#pragma unroll
    for (int rt = 0; rt < 2; ++rt) {
      int row = wr * 32 + rt * 16 + lr;
#pragma unroll
      for (int j = 0; j < 8; ++j) {
        int kk = lh * 16 + j * 2;
        a[rt].u[j] = Alds32[row * 20 + (kk >> 1)];
      }
    }
#pragma unroll
    for (int ct = 0; ct < 3; ++ct) {
      int col = wc * 48 + ct * 16 + lr;
#pragma unroll
      for (int j = 0; j < 8; ++j) b[ct].u[j] = Bs[(lh * 8 + j) * 96 + col];
    }
#pragma unroll
    for (int rt = 0; rt < 2; ++rt)
#pragma unroll
      for (int ct = 0; ct < 3; ++ct) acc[rt][ct] = wmma_bf16(a[rt], b[ct], acc[rt][ct]);
    __syncthreads();
  }

#pragma unroll
  for (int rt = 0; rt < 2; ++rt)
#pragma unroll
    for (int ct = 0; ct < 3; ++ct) {
      const int gcol = wc * 48 + ct * 16 + lr;
      if (gcol < HD) {
#pragma unroll
        for (int i = 0; i < 8; ++i) {
          const int grow = rowBase + wr * 32 + rt * 16 + lh * 8 + i;
          otmp[(size_t)(bq * SEQ + grow) * DIM + h * HD + gcol] = bf1(acc[rt][ct][i]);
        }
      }
    }
}

// ---------------- launch ----------------
extern "C" void kernel_launch(void* const* d_in, const int* in_sizes, int n_in,
                              void* d_out, int out_size, void* d_ws, size_t ws_size,
                              hipStream_t stream) {
  const float* hidden = (const float*)d_in[0];
  const float* qw = (const float*)d_in[1];
  const float* qb = (const float*)d_in[2];
  const float* kw = (const float*)d_in[3];
  const float* kb = (const float*)d_in[4];
  const float* vw = (const float*)d_in[5];
  const float* vb = (const float*)d_in[6];
  const float* ow = (const float*)d_in[7];
  const float* ob = (const float*)d_in[8];

  char* ws = (char*)d_ws;
  // workspace layout (bytes)
  unsigned short* hidb = (unsigned short*)ws;                 // 8192*1152*2 = 18,874,368
  unsigned* wq = (unsigned*)(ws + 18874368);                  // 4 x 576*1152*4
  unsigned* wk = wq + 663552;
  unsigned* wv = wk + 663552;
  unsigned* wo = wv + 663552;
  unsigned short* qp = (unsigned short*)(ws + 29491200);      // 25,165,824
  unsigned* kT = (unsigned*)(ws + 54657024);                  // 25,165,824
  unsigned* vp = (unsigned*)(ws + 79822848);                  // 25,165,824
  unsigned short* otmp = (unsigned short*)(ws + 104988672);   // 18,874,368 -> end 123,863,040

  float* outO = (float*)d_out;                                // [8,1024,1152]
  float* probs = outO + (size_t)MROWS * DIM;                  // [8,16,1024,1024]

  cvt_pairs<<<4096, 256, 0, stream>>>(hidden, (unsigned*)hidb, MROWS * DIM / 2);
  cvt_weight_pk<<<2048, 256, 0, stream>>>(qw, wq);
  cvt_weight_pk<<<2048, 256, 0, stream>>>(kw, wk);
  cvt_weight_pk<<<2048, 256, 0, stream>>>(vw, wv);
  cvt_weight_pk<<<2048, 256, 0, stream>>>(ow, wo);
  // zero padded q/k/v region (pad lanes must be 0 for correct WMMA)
  zero_u32<<<8192, 256, 0, stream>>>((unsigned*)qp, (size_t)(3 * 25165824) / 4);

  dim3 gProj(DIM / 128, MROWS / 128);  // (9, 64)
  proj_gemm<0><<<gProj, 256, 0, stream>>>(hidb, wq, qb, qp, nullptr);
  proj_gemm<1><<<gProj, 256, 0, stream>>>(hidb, wk, kb, (unsigned short*)kT, nullptr);
  proj_gemm<2><<<gProj, 256, 0, stream>>>(hidb, wv, vb, (unsigned short*)vp, nullptr);

  scores_gemm<<<dim3(SEQ / 128, SEQ / 128, BQ * NH), 256, 0, stream>>>(qp, kT, probs);
  softmax_rows<<<BQ * NH * SEQ, 256, 0, stream>>>(probs);
  attnv_gemm<<<dim3(SEQ / 128, BQ * NH), 256, 0, stream>>>(probs, vp, otmp);
  proj_gemm<3><<<gProj, 256, 0, stream>>>(otmp, wo, ob, nullptr, outO);
}
